// GPT_1236950581801
// MI455X (gfx1250) — compile-verified, matrix operand order checked by reading, not run on hardware
//
#include <hip/hip_runtime.h>
#include <hip/hip_bf16.h>

// ---------------------------------------------------------------------------
// GPT-2 (124M) forward for gfx1250 (MI455X).  bf16 WMMA + TDM tile staging.
//   L=12 H=12 D=768 HD=64 V=50257 B=2 T=1024  (BT=2048)
// ---------------------------------------------------------------------------

#define BT    2048
#define DMODEL 768
#define NHEAD  12
#define HDIM   64
#define NLAYER 12
#define VOCAB  50257
#define DFF    3072
#define SEQ    1024

typedef __attribute__((ext_vector_type(16))) __bf16 bf16x16;
typedef __attribute__((ext_vector_type(8)))  __bf16 bf16x8;
typedef __attribute__((ext_vector_type(8)))  float  floatx8;
typedef __attribute__((ext_vector_type(4)))  unsigned int uint32x4;
typedef __attribute__((ext_vector_type(8)))  int  int32x8;
typedef __attribute__((ext_vector_type(4)))  int  int32x4;

#if defined(__has_builtin)
#if __has_builtin(__builtin_amdgcn_tensor_load_to_lds) && \
    __has_builtin(__builtin_amdgcn_s_wait_tensorcnt)
#define USE_TDM 1
#endif
#endif
#ifndef USE_TDM
#define USE_TDM 0
#endif

// ---- bf16 helpers (round-to-nearest-even), raw ushort storage -------------
__device__ __forceinline__ unsigned short f2bf(float f) {
    unsigned int u = __float_as_uint(f);
    u += 0x7FFFu + ((u >> 16) & 1u);
    return (unsigned short)(u >> 16);
}
__device__ __forceinline__ float bf2f(unsigned short h) {
    return __uint_as_float(((unsigned int)h) << 16);
}
__device__ __forceinline__ float gelu_f(float x) {
    const float kAlpha = 0.7978845608028654f; // sqrt(2/pi)
    float t = tanhf(kAlpha * (x + 0.044715f * x * x * x));
    return 0.5f * x * (1.0f + t);
}

// ---- fragment load: two contiguous 16B chunks -> v16bf --------------------
__device__ __forceinline__ bf16x16 load_frag(const unsigned short* p0,
                                             const unsigned short* p1) {
    union { bf16x16 v; bf16x8 h[2]; } u;
    u.h[0] = *(const bf16x8*)p0;
    u.h[1] = *(const bf16x8*)p1;
    return u.v;
}
__device__ __forceinline__ floatx8 wmma_bf16(bf16x16 a, bf16x16 b, floatx8 c) {
    return __builtin_amdgcn_wmma_f32_16x16x32_bf16(
        false, a, false, b, (short)0, c, false, false);
}

// ===========================================================================
// Embedding: x[b,t,:] = wte[idx[b,t],:] + wpe[t,:]
// ===========================================================================
__global__ __launch_bounds__(256)
void embed_kernel(const int* __restrict__ idx, const float* __restrict__ wte,
                  const float* __restrict__ wpe, float* __restrict__ x) {
    const int tok = blockIdx.x;
    const int t   = tok & (SEQ - 1);
    const int id  = idx[tok];
    const float* wt = wte + (size_t)id * DMODEL;
    const float* wp = wpe + (size_t)t * DMODEL;
    float* xp = x + (size_t)tok * DMODEL;
    for (int c = threadIdx.x; c < DMODEL; c += 256) xp[c] = wt[c] + wp[c];
}

// ===========================================================================
// LayerNorm -> bf16 (one 256-thread block per token)
// ===========================================================================
__global__ __launch_bounds__(256)
void layernorm_kernel(const float* __restrict__ x, const float* __restrict__ w,
                      const float* __restrict__ b, unsigned short* __restrict__ out) {
    __shared__ float rs[8], rs2[8];
    __shared__ float s_mu, s_rstd;
    const int tok = blockIdx.x;
    const float* xp = x + (size_t)tok * DMODEL;
    float s = 0.f, s2 = 0.f;
    for (int c = threadIdx.x; c < DMODEL; c += 256) {
        float v = xp[c]; s += v; s2 += v * v;
    }
    for (int off = 16; off; off >>= 1) {
        s  += __shfl_down(s,  off, 32);
        s2 += __shfl_down(s2, off, 32);
    }
    if ((threadIdx.x & 31) == 0) { rs[threadIdx.x >> 5] = s; rs2[threadIdx.x >> 5] = s2; }
    __syncthreads();
    if (threadIdx.x == 0) {
        float a = 0.f, a2 = 0.f;
        for (int i = 0; i < 8; ++i) { a += rs[i]; a2 += rs2[i]; }
        float mu  = a  * (1.0f / DMODEL);
        float var = a2 * (1.0f / DMODEL) - mu * mu;
        s_mu = mu; s_rstd = rsqrtf(var + 1e-5f);
    }
    __syncthreads();
    const float mu = s_mu, rstd = s_rstd;
    unsigned short* op = out + (size_t)tok * DMODEL;
    for (int c = threadIdx.x; c < DMODEL; c += 256)
        op[c] = f2bf((xp[c] - mu) * rstd * w[c] + b[c]);
}

// ===========================================================================
// GEMM: C[M,N] = A_bf16[M,K] @ W_f32->bf16[K,N]  (+bias, + epilogue)
//   MODE 0: out_bf16 = acc + bias                     (QKV)
//   MODE 1: out_bf16 = gelu(acc + bias)               (FC1)
//   MODE 2: out_f32 += acc + bias                     (proj / FC2 residual)
//   MODE 3: out_f32  = acc          (no bias, N-guard) (lm_head)
// Block tile 128x128x32, 8 waves, wave tile 32x64 (2x4 WMMA accumulators).
// A tile staged by the Tensor Data Mover (pad 16dw+4dw -> 40-elem LDS stride).
// ===========================================================================
#define BM 128
#define BN 128
#define BK 32
#define LSTR 40   // LDS row stride (bf16 elems): 16B aligned, conflict-free

template <int MODE>
__global__ __launch_bounds__(256)
void gemm_kernel(const unsigned short* __restrict__ A, const float* __restrict__ W,
                 const float* __restrict__ bias, float* __restrict__ outf,
                 unsigned short* __restrict__ outb, int M, int N, int K) {
    __shared__ __align__(16) unsigned short sA[BM * LSTR];
    __shared__ __align__(16) unsigned short sB[BN * LSTR];

    const int tid  = threadIdx.x;
    const int lane = tid & 31;
    const int wave = tid >> 5;
    const int m0 = blockIdx.y * BM;
    const int n0 = blockIdx.x * BN;
    const int wm = (wave & 3) * 32;   // wave row offset within block tile
    const int wn = (wave >> 2) * 64;  // wave col offset within block tile
    const int lm = lane & 15;         // lane -> M (A) / N (B) index
    const int co = 8 * (lane >> 4);   // per-half K sub-offset

    floatx8 acc[2][4];
#pragma unroll
    for (int i = 0; i < 2; ++i)
#pragma unroll
        for (int j = 0; j < 4; ++j)
#pragma unroll
            for (int r = 0; r < 8; ++r) acc[i][j][r] = 0.0f;

    // staging roles
    const int brow = tid >> 3, bseg = (tid & 7) * 16;  // B: 8 thr/row, 16 f32 each

    for (int kt = 0; kt < K; kt += BK) {
        // ---- stage A tile (bf16 [128][32] -> LDS, 40-elem padded stride) --
#if USE_TDM
        if (tid < 32) {  // one wave drives the TDM (EXEC is ignored by TDM)
            const unsigned lds_base =
                (unsigned)(size_t)(&sA[0]);   // flat LDS addr: low 32 bits = LDS offset
            const unsigned long long ga =
                (unsigned long long)(size_t)(A + (size_t)m0 * K + kt);
            uint32x4 g0;
            g0[0] = 1u;                                  // count=1 (valid user D#)
            g0[1] = lds_base;                            // lds_addr
            g0[2] = (unsigned)(ga & 0xFFFFFFFFu);        // global_addr[31:0]
            g0[3] = (unsigned)((ga >> 32) & 0x1FFFFFFu)  // global_addr[56:32]
                    | 0x80000000u;                       // type=2 ("image")
            int32x8 g1;
            g1[0] = (int)0x06D10000;   // data_size=2B | pad_en | intv=16dw | amt=4dw
            g1[1] = (int)(32u  << 16); // tensor_dim0 = 32   (bits 63:48)
            g1[2] = (int)(128u << 16); // tensor_dim1 = 128  (bits 95:80)
            g1[3] = (int)(32u  << 16); // tile_dim0   = 32   (bits 127:112)
            g1[4] = (int)(128u);       // tile_dim1   = 128  (bits 143:128)
            g1[5] = (int)K;            // tensor_dim0_stride = K (row pitch, elems)
            g1[6] = 0;
            g1[7] = 0;
            int32x4 z4; z4[0] = z4[1] = z4[2] = z4[3] = 0;   // no group2/3 (2D)
            int32x8 z8;
#pragma unroll
            for (int i = 0; i < 8; ++i) z8[i] = 0;
            __builtin_amdgcn_tensor_load_to_lds(g0, g1, z4, z4, z8, 0);
        }
#else
        {
            const int arow = tid >> 1, aseg = (tid & 1) * 16;
            const uint4* ag = (const uint4*)(A + (size_t)(m0 + arow) * K + kt + aseg);
            uint4 v0 = ag[0], v1 = ag[1];
            uint4* as = (uint4*)&sA[arow * LSTR + aseg];
            as[0] = v0; as[1] = v1;
        }
#endif
        // ---- stage B tile: f32 -> bf16, transposed to [N=128][K=32] ------
        {
            const float* wg = W + (size_t)(kt + brow) * N + n0 + bseg;
            const int rem = N - (n0 + bseg);
            float vals[16];
            if (rem >= 16 && ((((size_t)wg) & 15) == 0)) {
                const float4* w4 = (const float4*)wg;
#pragma unroll
                for (int q = 0; q < 4; ++q) {
                    float4 f = w4[q];
                    vals[q * 4 + 0] = f.x; vals[q * 4 + 1] = f.y;
                    vals[q * 4 + 2] = f.z; vals[q * 4 + 3] = f.w;
                }
            } else {
#pragma unroll
                for (int j = 0; j < 16; ++j) vals[j] = (j < rem) ? wg[j] : 0.0f;
            }
#pragma unroll
            for (int j = 0; j < 16; ++j)
                sB[(bseg + j) * LSTR + brow] = f2bf(vals[j]);
        }
        if (kt + BK < K) {  // prefetch next weight tile -> global_prefetch_b8
            __builtin_prefetch(W + (size_t)(kt + BK + brow) * N + n0 + bseg, 0, 1);
        }
#if USE_TDM
        if (tid < 32) __builtin_amdgcn_s_wait_tensorcnt(0);
#endif
        __syncthreads();

        // ---- WMMA over the staged tile -----------------------------------
        bf16x16 af[2], bfr[4];
#pragma unroll
        for (int i = 0; i < 2; ++i) {
            const unsigned short* p = &sA[(wm + i * 16 + lm) * LSTR];
            af[i] = load_frag(p + co, p + 16 + co);
        }
#pragma unroll
        for (int j = 0; j < 4; ++j) {
            const unsigned short* p = &sB[(wn + j * 16 + lm) * LSTR];
            bfr[j] = load_frag(p + co, p + 16 + co);
        }
#pragma unroll
        for (int i = 0; i < 2; ++i)
#pragma unroll
            for (int j = 0; j < 4; ++j)
                acc[i][j] = wmma_bf16(af[i], bfr[j], acc[i][j]);
        __syncthreads();
    }

    // ---- epilogue (C layout: lane = N, VGPR r = row r / r+8 per half) ----
#pragma unroll
    for (int j = 0; j < 4; ++j) {
        const int col = n0 + wn + j * 16 + lm;
        float bval = 0.0f;
        if (MODE != 3) bval = bias[col];
#pragma unroll
        for (int i = 0; i < 2; ++i) {
#pragma unroll
            for (int r = 0; r < 8; ++r) {
                const int row = m0 + wm + i * 16 + r + co;
                float v = acc[i][j][r] + bval;
                if (MODE == 0) {
                    outb[(size_t)row * N + col] = f2bf(v);
                } else if (MODE == 1) {
                    outb[(size_t)row * N + col] = f2bf(gelu_f(v));
                } else if (MODE == 2) {
                    size_t o = (size_t)row * N + col;
                    outf[o] += v;
                } else {
                    if (col < N) outf[(size_t)row * N + col] = v;
                }
            }
        }
    }
}

// ===========================================================================
// Split QKV (bf16 [BT][3D]) into per-head q,k: [B,H,T,HD]; vT: [B,H,HD,T]
// q is pre-scaled by 1/sqrt(HD).
// ===========================================================================
__global__ __launch_bounds__(256)
void split_qkv_kernel(const unsigned short* __restrict__ qkv,
                      unsigned short* __restrict__ q, unsigned short* __restrict__ k,
                      unsigned short* __restrict__ vT) {
    const int tok = blockIdx.x;
    const int b = tok >> 10, t = tok & (SEQ - 1);
    const unsigned short* src = qkv + (size_t)tok * (3 * DMODEL);
    for (int c = threadIdx.x; c < 3 * DMODEL; c += 256) {
        const unsigned short raw = src[c];
        const int s  = c / DMODEL;
        const int cc = c - s * DMODEL;
        const int h  = cc >> 6, hd = cc & 63;
        const int bh = b * NHEAD + h;
        if (s == 0) {
            q[((size_t)bh * SEQ + t) * HDIM + hd] = f2bf(bf2f(raw) * 0.125f);
        } else if (s == 1) {
            k[((size_t)bh * SEQ + t) * HDIM + hd] = raw;
        } else {
            vT[((size_t)bh * HDIM + hd) * SEQ + t] = raw;
        }
    }
}

// ===========================================================================
// Flash attention: block = 4 waves, each wave owns a 16-row q tile.
// grid = (T/64, B*H).  Streaming softmax, WMMA for S=q@k^T and O=P@V.
// Causal masking applied only on boundary key blocks.
// ===========================================================================
__global__ __launch_bounds__(128)
void attention_kernel(const unsigned short* __restrict__ q,
                      const unsigned short* __restrict__ k,
                      const unsigned short* __restrict__ vT,
                      unsigned short* __restrict__ y) {
    __shared__ __align__(16) unsigned short sP[4][16 * LSTR]; // per-wave P scratch
    const int lane = threadIdx.x & 31;
    const int wave = threadIdx.x >> 5;
    const int bh = blockIdx.y;
    const int b = bh / NHEAD, h = bh % NHEAD;
    const int qbase = blockIdx.x * 64 + wave * 16;
    const int lm = lane & 15;
    const int co = 8 * (lane >> 4);

    const unsigned short* qp = q  + (size_t)bh * SEQ * HDIM;
    const unsigned short* kp = k  + (size_t)bh * SEQ * HDIM;
    const unsigned short* vp = vT + (size_t)bh * HDIM * SEQ;

    // q fragments for this wave's 16 rows (HD=64 -> 2 K-steps of 32)
    bf16x16 aq[2];
    {
        const unsigned short* qr = qp + (size_t)(qbase + lm) * HDIM;
        aq[0] = load_frag(qr + co,      qr + 16 + co);
        aq[1] = load_frag(qr + 32 + co, qr + 48 + co);
    }

    floatx8 o[4];
    float mrow[8], lrow[8];
#pragma unroll
    for (int j = 0; j < 4; ++j)
#pragma unroll
        for (int r = 0; r < 8; ++r) o[j][r] = 0.0f;
#pragma unroll
    for (int r = 0; r < 8; ++r) { mrow[r] = -1e30f; lrow[r] = 0.0f; }

    unsigned short* myP = &sP[wave][0];

    for (int kb = 0; kb <= qbase + 15; kb += 32) {
        // ---- S = q @ k^T for 32 keys (two 16-wide tiles) ----------------
        floatx8 s0, s1;
#pragma unroll
        for (int r = 0; r < 8; ++r) { s0[r] = 0.0f; s1[r] = 0.0f; }
        {
            const unsigned short* kr0 = kp + (size_t)(kb + lm) * HDIM;
            const unsigned short* kr1 = kp + (size_t)(kb + 16 + lm) * HDIM;
            bf16x16 bk;
            bk = load_frag(kr0 + co,      kr0 + 16 + co); s0 = wmma_bf16(aq[0], bk, s0);
            bk = load_frag(kr0 + 32 + co, kr0 + 48 + co); s0 = wmma_bf16(aq[1], bk, s0);
            bk = load_frag(kr1 + co,      kr1 + 16 + co); s1 = wmma_bf16(aq[0], bk, s1);
            bk = load_frag(kr1 + 32 + co, kr1 + 48 + co); s1 = wmma_bf16(aq[1], bk, s1);
        }
        // ---- causal mask only where the block crosses the diagonal ------
        if (kb + 31 > qbase) {
#pragma unroll
            for (int r = 0; r < 8; ++r) {
                const int qa = qbase + r + co;
                s0[r] = (kb + lm      <= qa) ? s0[r] : -1e30f;
                s1[r] = (kb + 16 + lm <= qa) ? s1[r] : -1e30f;
            }
        }
        // ---- streaming softmax ------------------------------------------
        float mnew[8];
#pragma unroll
        for (int r = 0; r < 8; ++r) {
            float mx = fmaxf(s0[r], s1[r]);
#pragma unroll
            for (int d = 1; d < 16; d <<= 1) mx = fmaxf(mx, __shfl_xor(mx, d, 32));
            mnew[r] = fmaxf(mrow[r], mx);
        }
#pragma unroll
        for (int r = 0; r < 8; ++r) {
            const float corr = expf(mrow[r] - mnew[r]);
            const float p0 = expf(s0[r] - mnew[r]);
            const float p1 = expf(s1[r] - mnew[r]);
            float ps = p0 + p1;
#pragma unroll
            for (int d = 1; d < 16; d <<= 1) ps += __shfl_xor(ps, d, 32);
            lrow[r] = lrow[r] * corr + ps;
            mrow[r] = mnew[r];
#pragma unroll
            for (int j = 0; j < 4; ++j) o[j][r] *= corr;
            myP[(r + co) * LSTR + lm]      = f2bf(p0);
            myP[(r + co) * LSTR + 16 + lm] = f2bf(p1);
        }
        __builtin_amdgcn_wave_barrier();  // keep P stores before P loads
        // ---- O += P @ V -------------------------------------------------
        bf16x16 pa = load_frag(&myP[lm * LSTR + co], &myP[lm * LSTR + 16 + co]);
#pragma unroll
        for (int j = 0; j < 4; ++j) {
            const unsigned short* vr = vp + (size_t)(j * 16 + lm) * SEQ + kb;
            bf16x16 bv = load_frag(vr + co, vr + 16 + co);
            o[j] = wmma_bf16(pa, bv, o[j]);
        }
    }

    // ---- normalize and store y[token][h*64 + hd] ------------------------
#pragma unroll
    for (int j = 0; j < 4; ++j)
#pragma unroll
        for (int r = 0; r < 8; ++r) {
            const int row = b * SEQ + qbase + r + co;
            y[(size_t)row * DMODEL + h * HDIM + j * 16 + lm] = f2bf(o[j][r] / lrow[r]);
        }
}

// ===========================================================================
// Host-side orchestration
// ===========================================================================
extern "C" void kernel_launch(void* const* d_in, const int* in_sizes, int n_in,
                              void* d_out, int out_size, void* d_ws, size_t ws_size,
                              hipStream_t stream) {
    (void)in_sizes; (void)n_in; (void)out_size; (void)ws_size;
    const int*   idx    = (const int*)d_in[0];
    const float* wte    = (const float*)d_in[1];
    const float* wpe    = (const float*)d_in[2];
    const float* ln1_w  = (const float*)d_in[3];
    const float* ln1_b  = (const float*)d_in[4];
    const float* attn_w = (const float*)d_in[5];
    const float* attn_b = (const float*)d_in[6];
    const float* proj_w = (const float*)d_in[7];
    const float* proj_b = (const float*)d_in[8];
    const float* ln2_w  = (const float*)d_in[9];
    const float* ln2_b  = (const float*)d_in[10];
    const float* fc_w   = (const float*)d_in[11];
    const float* fc_b   = (const float*)d_in[12];
    const float* fc2_w  = (const float*)d_in[13];
    const float* fc2_b  = (const float*)d_in[14];
    const float* lnf_w  = (const float*)d_in[15];
    const float* lnf_b  = (const float*)d_in[16];
    const float* lm_w   = (const float*)d_in[17];
    float* out = (float*)d_out;

    char* ws = (char*)d_ws;
    float*          x    = (float*)(ws);                       //  6.0 MB f32 [BT][D]
    unsigned short* hbuf = (unsigned short*)(ws + 6291456);    //  3.0 MB bf16 [BT][D]
    unsigned short* qkv  = (unsigned short*)(ws + 9437184);    //  9.0 MB bf16 [BT][3D]
    unsigned short* qh   = (unsigned short*)(ws + 18874368);   //  3.0 MB [B,H,T,HD]
    unsigned short* kh   = (unsigned short*)(ws + 22020096);   //  3.0 MB [B,H,T,HD]
    unsigned short* vT   = (unsigned short*)(ws + 25165824);   //  3.0 MB [B,H,HD,T]
    unsigned short* ybuf = (unsigned short*)(ws + 28311552);   //  3.0 MB bf16 [BT][D]
    unsigned short* mbuf = (unsigned short*)(ws + 31457280);   // 12.0 MB bf16 [BT][4D]

    embed_kernel<<<BT, 256, 0, stream>>>(idx, wte, wpe, x);

    for (int l = 0; l < NLAYER; ++l) {
        layernorm_kernel<<<BT, 256, 0, stream>>>(x, ln1_w + l * DMODEL,
                                                 ln1_b + l * DMODEL, hbuf);
        gemm_kernel<0><<<dim3(18, 16), 256, 0, stream>>>(
            hbuf, attn_w + (size_t)l * DMODEL * 3 * DMODEL, attn_b + l * 3 * DMODEL,
            nullptr, qkv, BT, 3 * DMODEL, DMODEL);
        split_qkv_kernel<<<BT, 256, 0, stream>>>(qkv, qh, kh, vT);
        attention_kernel<<<dim3(SEQ / 64, 2 * NHEAD), 128, 0, stream>>>(qh, kh, vT, ybuf);
        gemm_kernel<2><<<dim3(6, 16), 256, 0, stream>>>(
            ybuf, proj_w + (size_t)l * DMODEL * DMODEL, proj_b + l * DMODEL,
            x, nullptr, BT, DMODEL, DMODEL);
        layernorm_kernel<<<BT, 256, 0, stream>>>(x, ln2_w + l * DMODEL,
                                                 ln2_b + l * DMODEL, hbuf);
        gemm_kernel<1><<<dim3(24, 16), 256, 0, stream>>>(
            hbuf, fc_w + (size_t)l * DMODEL * DFF, fc_b + l * DFF,
            nullptr, mbuf, BT, DFF, DMODEL);
        gemm_kernel<2><<<dim3(6, 16), 256, 0, stream>>>(
            mbuf, fc2_w + (size_t)l * DFF * DMODEL, fc2_b + l * DMODEL,
            x, nullptr, BT, DMODEL, DFF);
    }

    layernorm_kernel<<<BT, 256, 0, stream>>>(x, lnf_w, lnf_b, hbuf);
    gemm_kernel<3><<<dim3((VOCAB + BN - 1) / BN, 16), 256, 0, stream>>>(
        hbuf, lm_w, nullptr, out, nullptr, BT, VOCAB, DMODEL);
}
